// EquivariantMessagePassing_65532611002569
// MI455X (gfx1250) — matrix operand order repaired and checked by reading, not compiled, not run on hardware
//
#include <hip/hip_runtime.h>
#include <hip/hip_bf16.h>

#define NNODES 50000
#define NEDGES 800000
#define NATTR  32
#define FEAT   240

typedef __attribute__((ext_vector_type(16))) __bf16       v16bf;
typedef __attribute__((ext_vector_type(8)))  float        v8f;
typedef __attribute__((ext_vector_type(8)))  unsigned int v8u;

// ws layout (dwords): [0,65536) W0 bf16-packed; [65536,81920) W1; [81920,86016) W2;
// [86016, 86016+50000) counts
#define W0_OFF 0
#define W1_OFF 65536
#define W2_OFF 81920
#define CNT_OFF 86016

__device__ __forceinline__ unsigned int pack2_bf16(float lo, float hi) {
  unsigned a = __float_as_uint(lo); a += 0x7FFFu + ((a >> 16) & 1u);   // RNE
  unsigned b = __float_as_uint(hi); b += 0x7FFFu + ((b >> 16) & 1u);
  return (a >> 16) | (b & 0xFFFF0000u);
}

// ---------------- zero out + counts ----------------
__global__ void emp_zero(float* __restrict__ out, float* __restrict__ counts) {
  long long i = (long long)blockIdx.x * blockDim.x + threadIdx.x;
  if (i < (long long)NNODES * FEAT) out[i] = 0.0f;
  else if (i < (long long)NNODES * FEAT + NNODES) counts[i - (long long)NNODES * FEAT] = 0.0f;
}

// ---------------- pack weights to bf16 in WMMA-B register layout ----------------
// B (32x16, bf16): VGPR j<4: lanes 0-15 hold K=2j,2j+1, lanes 16-31 K=8+2j,8+2j+1; j>=4: +8 more (lanes 0-15 K=8+2j.., lanes 16-31 K=16+2j..).
// Stored lane-major per 16-col tile: dword index = base + t*256 + lane*8 + j.
__global__ void emp_prep(const float* __restrict__ w0, const float* __restrict__ w1,
                         const float* __restrict__ w2, unsigned int* __restrict__ wpk) {
  int dw = blockIdx.x * 256 + threadIdx.x;          // < 86016
  const float* w; int mul, sh, local; float norm;
  if (dw < W1_OFF)      { w = w0; mul = 64; sh = 6; local = dw - W0_OFF; norm = 0.02209708691f; } // 1/sqrt(64*32)
  else if (dw < W2_OFF) { w = w1; mul = 32; sh = 5; local = dw - W1_OFF; norm = 0.03125f;       } // 1/sqrt(32*32)
  else                  { w = w2; mul = 16; sh = 4; local = dw - W2_OFF; norm = 0.04419417382f; } // 1/sqrt(16*32)
  int t    = local >> 8;
  int wi   = local & 255;
  int lane = wi >> 3;
  int j    = wi & 7;
  int c = t * 16 + (lane & 15);       // flattened column = m*mul + o
  int m = c >> sh;
  int o = c & (mul - 1);
  int nlo = (lane >> 4) * 8 + 2 * j + ((j >= 4) ? 8 : 0);
  float lo = w[((size_t)m * NATTR + nlo)     * mul + o] * norm;
  float hi = w[((size_t)m * NATTR + nlo + 1) * mul + o] * norm;
  wpk[dw] = pack2_bf16(lo, hi);
}

// ---------------- main fused kernel: 1 wave = 16 edges ----------------
__global__ __launch_bounds__(128)
void emp_main(const float* __restrict__ x, const long long* __restrict__ ei,
              const float* __restrict__ attrs, const unsigned int* __restrict__ wpk,
              float* __restrict__ out, float* __restrict__ counts) {
  __shared__ float xsh[4][16][241];   // staged gathered node features (241: conflict-free stride)
  __shared__ int   dsh[4][16];

  const int w    = threadIdx.x >> 5;
  const int lane = threadIdx.x & 31;
  const int row  = lane & 15;
  const int half = lane >> 4;
  const int e    = (blockIdx.x * 4 + w) * 16 + row;

  const int srcn = (int)ei[e];
  const int dstn = (int)ei[(long long)NEDGES + e];
  if (half == 0) dsh[w][row] = dstn;

  // Stage x[src] for this wave's 16 edges: each lane-half loads 120 floats of its row.
  {
    const float* xr = x + (size_t)srcn * FEAT + half * 120;
    #pragma unroll
    for (int k = 0; k < 120; k += 4) {
      float4 v = *(const float4*)(xr + k);
      int c = half * 120 + k;
      xsh[w][row][c]     = v.x; xsh[w][row][c + 1] = v.y;
      xsh[w][row][c + 2] = v.z; xsh[w][row][c + 3] = v.w;
    }
  }

  // A matrix (16 edges x 32 attrs, bf16): lane-half 0 -> K {0..7,16..23}; half 1 -> K {8..15,24..31}
  const float* ar = attrs + (size_t)e * NATTR + half * 8;
  float4 p0 = *(const float4*)(ar);
  float4 p1 = *(const float4*)(ar + 4);
  float4 p2 = *(const float4*)(ar + 16);
  float4 p3 = *(const float4*)(ar + 20);
  v8u au;
  au[0] = pack2_bf16(p0.x, p0.y); au[1] = pack2_bf16(p0.z, p0.w);
  au[2] = pack2_bf16(p1.x, p1.y); au[3] = pack2_bf16(p1.z, p1.w);
  au[4] = pack2_bf16(p2.x, p2.y); au[5] = pack2_bf16(p2.z, p2.w);
  au[6] = pack2_bf16(p3.x, p3.y); au[7] = pack2_bf16(p3.z, p3.w);
  v16bf amat = __builtin_bit_cast(v16bf, au);

  __syncthreads();

  const unsigned int* W0 = wpk + W0_OFF;
  const unsigned int* W1 = wpk + W1_OFF;
  const unsigned int* W2 = wpk + W2_OFF;
  const v8f zero = {};

  // ======== block 0: mul=64, l=0 (d=1). cols c=m*64+o -> tile t=m*4+ot holds one m. ========
  float acc0[4][8];
  #pragma unroll
  for (int ot = 0; ot < 4; ++ot)
    #pragma unroll
    for (int r = 0; r < 8; ++r) acc0[ot][r] = 0.0f;

  for (int m = 0; m < 64; ++m) {
    float xm[8];
    #pragma unroll
    for (int r = 0; r < 8; ++r) xm[r] = xsh[w][r + 8 * half][m];
    // WGP-scope prefetch (locality 3): weights are hot for every wave on the WGP
    if (m < 63) __builtin_prefetch(W0 + (size_t)((m + 1) * 4) * 256 + lane * 8, 0, 3);
    #pragma unroll
    for (int ot = 0; ot < 4; ++ot) {
      v16bf bmat = __builtin_bit_cast(v16bf, *(const v8u*)(W0 + (size_t)(m * 4 + ot) * 256 + lane * 8));
      v8f d = __builtin_amdgcn_wmma_f32_16x16x32_bf16(false, amat, false, bmat, (short)0, zero, false, false);
      #pragma unroll
      for (int r = 0; r < 8; ++r) acc0[ot][r] += d[r] * xm[r];
    }
  }

  // ======== block 1: mul=32, l=1 (d=3). tile t=m*2+ot. x at 64+m*3+i. ========
  float acc1[2][3][8];
  #pragma unroll
  for (int ot = 0; ot < 2; ++ot)
    #pragma unroll
    for (int i = 0; i < 3; ++i)
      #pragma unroll
      for (int r = 0; r < 8; ++r) acc1[ot][i][r] = 0.0f;

  for (int m = 0; m < 32; ++m) {
    float xm[3][8];
    #pragma unroll
    for (int i = 0; i < 3; ++i)
      #pragma unroll
      for (int r = 0; r < 8; ++r) xm[i][r] = xsh[w][r + 8 * half][64 + m * 3 + i];
    if (m < 31) __builtin_prefetch(W1 + (size_t)((m + 1) * 2) * 256 + lane * 8, 0, 3);
    #pragma unroll
    for (int ot = 0; ot < 2; ++ot) {
      v16bf bmat = __builtin_bit_cast(v16bf, *(const v8u*)(W1 + (size_t)(m * 2 + ot) * 256 + lane * 8));
      v8f d = __builtin_amdgcn_wmma_f32_16x16x32_bf16(false, amat, false, bmat, (short)0, zero, false, false);
      #pragma unroll
      for (int i = 0; i < 3; ++i)
        #pragma unroll
        for (int r = 0; r < 8; ++r) acc1[ot][i][r] += d[r] * xm[i][r];
    }
  }

  // ======== block 2: mul=16, l=2 (d=5). tile t=m. x at 160+m*5+i. ========
  float acc2[5][8];
  #pragma unroll
  for (int i = 0; i < 5; ++i)
    #pragma unroll
    for (int r = 0; r < 8; ++r) acc2[i][r] = 0.0f;

  for (int m = 0; m < 16; ++m) {
    float xm[5][8];
    #pragma unroll
    for (int i = 0; i < 5; ++i)
      #pragma unroll
      for (int r = 0; r < 8; ++r) xm[i][r] = xsh[w][r + 8 * half][160 + m * 5 + i];
    v16bf bmat = __builtin_bit_cast(v16bf, *(const v8u*)(W2 + (size_t)m * 256 + lane * 8));
    v8f d = __builtin_amdgcn_wmma_f32_16x16x32_bf16(false, amat, false, bmat, (short)0, zero, false, false);
    #pragma unroll
    for (int i = 0; i < 5; ++i)
      #pragma unroll
      for (int r = 0; r < 8; ++r) acc2[i][r] += d[r] * xm[i][r];
  }

  // ======== scatter-add (output region is L2-resident; f32 atomics at L2) ========
  #pragma unroll
  for (int r = 0; r < 8; ++r) {
    const int   node = dsh[w][r + 8 * half];
    float* ob = out + (size_t)node * FEAT;
    #pragma unroll
    for (int ot = 0; ot < 4; ++ot)
      unsafeAtomicAdd(ob + ot * 16 + row, acc0[ot][r]);
    #pragma unroll
    for (int ot = 0; ot < 2; ++ot)
      #pragma unroll
      for (int i = 0; i < 3; ++i)
        unsafeAtomicAdd(ob + 64 + (ot * 16 + row) * 3 + i, acc1[ot][i][r]);
    #pragma unroll
    for (int i = 0; i < 5; ++i)
      unsafeAtomicAdd(ob + 160 + row * 5 + i, acc2[i][r]);
  }
  if (half == 0) unsafeAtomicAdd(counts + dstn, 1.0f);
}

// ---------------- finalize: mean ----------------
__global__ void emp_fin(float* __restrict__ out, const float* __restrict__ counts) {
  long long i = (long long)blockIdx.x * blockDim.x + threadIdx.x;
  if (i >= (long long)NNODES * FEAT) return;
  int n = (int)(i / FEAT);
  out[i] = out[i] / fmaxf(counts[n], 1.0f);
}

extern "C" void kernel_launch(void* const* d_in, const int* in_sizes, int n_in,
                              void* d_out, int out_size, void* d_ws, size_t ws_size,
                              hipStream_t stream) {
  const float*     x     = (const float*)d_in[0];
  const long long* ei    = (const long long*)d_in[1];   // int64 edge_index [2, E]
  const float*     attrs = (const float*)d_in[2];
  const float*     w0    = (const float*)d_in[3];
  const float*     w1    = (const float*)d_in[4];
  const float*     w2    = (const float*)d_in[5];
  float*        out  = (float*)d_out;
  unsigned int* wpk  = (unsigned int*)d_ws;
  float*        cnts = (float*)d_ws + CNT_OFF;

  {
    long long n = (long long)NNODES * FEAT + NNODES;
    emp_zero<<<(unsigned)((n + 255) / 256), 256, 0, stream>>>(out, cnts);
  }
  emp_prep<<<336, 256, 0, stream>>>(w0, w1, w2, wpk);           // 86016 packed dwords
  emp_main<<<NEDGES / 16 / 4, 128, 0, stream>>>(x, ei, attrs, wpk, out, cnts);
  {
    long long n = (long long)NNODES * FEAT;
    emp_fin<<<(unsigned)((n + 255) / 256), 256, 0, stream>>>(out, cnts);
  }
}
